// PointNetPlusPlus_15092515078589
// MI455X (gfx1250) — matrix operand order repaired and checked by reading, not compiled, not run on hardware
//
#include <hip/hip_runtime.h>

// PointNet++ forward for MI455X (gfx1250, wave32).
// GEMM-heavy MLP layers run on v_wmma_f32_16x16x32_f16 with a 16x64 tile per
// wave (4 independent accumulators) and a software-pipelined k-loop: k+1
// fragment loads are issued before the k-step WMMAs so VMEM latency hides
// behind the XDL pipe instead of draining loadcnt to 0 before every WMMA.
// Everything else (FPS, ball query, grouping, BN stats, pooling, FC head) is
// VALU/LDS work and is bandwidth/latency noise at 23.3 TB/s.

typedef __attribute__((ext_vector_type(16))) _Float16 v16h;
typedef __attribute__((ext_vector_type(8)))  _Float16 v8h;
typedef __attribute__((ext_vector_type(8)))  float    v8f;

#define BN_EPS 1e-5f

// ---------------- Farthest Point Sampling ----------------
// One block per batch. dmin lives in LDS; argmax with first-index tie-break
// to match jnp.argmax. Sequential over S samples (inherent to FPS).
__global__ void fps_kernel(const float* __restrict__ xyz, int* __restrict__ fidx,
                           int N, int S) {
  int b = blockIdx.x;
  __shared__ float dmin[4096];
  __shared__ float rv[256];
  __shared__ int   ri[256];
  int t = threadIdx.x;
  for (int i = t; i < N; i += 256) dmin[i] = 1e10f;
  if (t == 0) fidx[b * S] = 0;
  __syncthreads();
  int last = 0;
  for (int s = 1; s < S; ++s) {
    float px = xyz[((size_t)b * N + last) * 3 + 0];
    float py = xyz[((size_t)b * N + last) * 3 + 1];
    float pz = xyz[((size_t)b * N + last) * 3 + 2];
    float best = -1.0f; int bi = 0x7fffffff;
    for (int i = t; i < N; i += 256) {
      float dx = xyz[((size_t)b * N + i) * 3 + 0] - px;
      float dy = xyz[((size_t)b * N + i) * 3 + 1] - py;
      float dz = xyz[((size_t)b * N + i) * 3 + 2] - pz;
      float d  = dx * dx + dy * dy + dz * dz;
      float dm = fminf(dmin[i], d);
      dmin[i] = dm;
      if (dm > best) { best = dm; bi = i; }   // strict > keeps lowest index
    }
    rv[t] = best; ri[t] = bi;
    __syncthreads();
    for (int off = 128; off > 0; off >>= 1) {
      if (t < off) {
        if (rv[t + off] > rv[t] || (rv[t + off] == rv[t] && ri[t + off] < ri[t])) {
          rv[t] = rv[t + off]; ri[t] = ri[t + off];
        }
      }
      __syncthreads();
    }
    last = ri[0];
    if (t == 0) fidx[b * S + s] = last;
    __syncthreads();
  }
}

__global__ void gather_xyz_kernel(const float* __restrict__ xyz, const int* __restrict__ idx,
                                  float* __restrict__ out, int N, int S, int total) {
  int i = blockIdx.x * blockDim.x + threadIdx.x;
  if (i >= total) return;            // total = B*S, flat center index
  int b = i / S;
  int id = idx[i];
  out[i * 3 + 0] = xyz[((size_t)b * N + id) * 3 + 0];
  out[i * 3 + 1] = xyz[((size_t)b * N + id) * 3 + 1];
  out[i * 3 + 2] = xyz[((size_t)b * N + id) * 3 + 2];
}

// ---------------- Ball query ----------------
// One wave (32 lanes) per center. Ballot-based stream compaction selects the
// ns smallest in-radius indices in ascending order, padded with the first
// (exactly the reference's sort-and-pad semantics).
__global__ void ball_query_kernel(const float* __restrict__ xyz, const float* __restrict__ new_xyz,
                                  int* __restrict__ gidx, int N, int S, int ns, float r2) {
  int c = blockIdx.x;                // b*S + s
  int b = c / S;
  int l = threadIdx.x;               // 0..31
  float cx = new_xyz[c * 3 + 0], cy = new_xyz[c * 3 + 1], cz = new_xyz[c * 3 + 2];
  __shared__ int cnt;
  if (l == 0) cnt = 0;
  __syncthreads();
  for (int base = 0; base < N; base += 32) {
    int cbase = cnt;
    int i = base + l;
    float dx = xyz[((size_t)b * N + i) * 3 + 0] - cx;
    float dy = xyz[((size_t)b * N + i) * 3 + 1] - cy;
    float dz = xyz[((size_t)b * N + i) * 3 + 2] - cz;
    bool in = (dx * dx + dy * dy + dz * dz) <= r2;
    unsigned mask = (unsigned)__ballot(in);
    int prior = __popc(mask & ((1u << l) - 1u));
    int slot = cbase + prior;
    if (in && slot < ns) gidx[(size_t)c * ns + slot] = i;
    __syncthreads();
    if (l == 0) cnt = cbase + __popc(mask);
    __syncthreads();
    if (cnt >= ns) break;            // uniform across the wave
  }
  __syncthreads();
  int total = cnt < ns ? cnt : ns;   // >=1 (center itself is in radius)
  int first = gidx[(size_t)c * ns];
  for (int s2 = total + l; s2 < ns; s2 += 32) gidx[(size_t)c * ns + s2] = first;
}

// ---------------- Grouping: build f16 A matrices (K padded to mult of 32) ----
__global__ void group_sa1_kernel(const float* __restrict__ xyz, const float* __restrict__ new_xyz,
                                 const int* __restrict__ gidx, _Float16* __restrict__ A,
                                 int N, int S, int ns, int Kp, int rows) {
  int r = blockIdx.x * blockDim.x + threadIdx.x;
  if (r >= rows) return;
  int c = r / ns, j = r % ns;
  int b = c / S;
  int id = gidx[(size_t)c * ns + j];
  const float* p = &xyz[((size_t)b * N + id) * 3];
  _Float16* row = A + (size_t)r * Kp;
  row[0] = (_Float16)(p[0] - new_xyz[c * 3 + 0]);
  row[1] = (_Float16)(p[1] - new_xyz[c * 3 + 1]);
  row[2] = (_Float16)(p[2] - new_xyz[c * 3 + 2]);
  for (int k = 3; k < Kp; ++k) row[k] = (_Float16)0.f;
}

__global__ void group_sa2_kernel(const float* __restrict__ xyz, const float* __restrict__ new_xyz,
                                 const int* __restrict__ gidx, const float* __restrict__ feats,
                                 int C, _Float16* __restrict__ A,
                                 int N, int S, int ns, int Kp, int rows) {
  int r = blockIdx.x * blockDim.x + threadIdx.x;
  if (r >= rows) return;
  int c = r / ns, j = r % ns;
  int b = c / S;
  int id = gidx[(size_t)c * ns + j];
  const float* p = &xyz[((size_t)b * N + id) * 3];
  _Float16* row = A + (size_t)r * Kp;
  row[0] = (_Float16)(p[0] - new_xyz[c * 3 + 0]);
  row[1] = (_Float16)(p[1] - new_xyz[c * 3 + 1]);
  row[2] = (_Float16)(p[2] - new_xyz[c * 3 + 2]);
  const float* f = &feats[((size_t)b * N + id) * C];
  for (int k = 0; k < C; ++k) row[3 + k] = (_Float16)f[k];
  for (int k = 3 + C; k < Kp; ++k) row[k] = (_Float16)0.f;
}

__global__ void group_sa3_kernel(const float* __restrict__ xyz2, const float* __restrict__ f2,
                                 int C, _Float16* __restrict__ A, int Kp, int rows) {
  int r = blockIdx.x * blockDim.x + threadIdx.x;
  if (r >= rows) return;             // rows = B*128, row = (b,i)
  _Float16* row = A + (size_t)r * Kp;
  row[0] = (_Float16)xyz2[r * 3 + 0];
  row[1] = (_Float16)xyz2[r * 3 + 1];
  row[2] = (_Float16)xyz2[r * 3 + 2];
  const float* f = &f2[(size_t)r * C];
  for (int k = 0; k < C; ++k) row[3 + k] = (_Float16)f[k];
  for (int k = 3 + C; k < Kp; ++k) row[k] = (_Float16)0.f;
}

// ---------------- Weight pre-pack into WMMA B-fragment lane order ----------
// Element e of lane l for tile (kb, n0):  K = kb*32 + 8*(l>>4) + e + 8*(e>>3),
//                                         N = n0*16 + (l&15)
__global__ void pack_w_kernel(const float* __restrict__ W, _Float16* __restrict__ Wpk,
                              int K, int Kp, int N) {
  int i = blockIdx.x * blockDim.x + threadIdx.x;
  int total = Kp * N;
  if (i >= total) return;
  int e    = i & 15;
  int lane = (i >> 4) & 31;
  int t    = i >> 9;
  int nt   = N >> 4;
  int n0   = t % nt;
  int kb   = t / nt;
  int k = kb * 32 + 8 * (lane >> 4) + e + 8 * (e >> 3);
  int n = n0 * 16 + (lane & 15);
  float v = (k < K) ? W[(size_t)k * N + n] : 0.f;
  Wpk[i] = (_Float16)v;
}

__global__ void zero_kernel(float* __restrict__ p, int n) {
  int i = blockIdx.x * blockDim.x + threadIdx.x;
  if (i < n) p[i] = 0.f;
}

// ---------------- WMMA GEMM + bias + BN-stat epilogue ----------------------
// One wave per 16(M) x 64(N) output tile: one A-fragment feeds four
// v_wmma_f32_16x16x32_f16 with independent accumulators. The k-loop is
// software-pipelined: fragment loads for step k+1 are issued before the four
// WMMAs of step k, so s_wait_loadcnt lands after the WMMAs have issued.
// A: [M, Kp] f16 row-major. Wpk: packed B fragments. Y: [M, N] f32
// pre-activation. sums: [2N] (sum, sumsq) accumulated via f32 atomics.
__device__ __forceinline__ v16h load_a_frag(const _Float16* __restrict__ arow, int kb) {
  v8h lo = *(const v8h*)(arow + (size_t)kb * 32);
  v8h hi = *(const v8h*)(arow + (size_t)kb * 32 + 16);
  v16h a;
#pragma unroll
  for (int i = 0; i < 8; ++i) { a[i] = lo[i]; a[i + 8] = hi[i]; }
  return a;
}

__global__ void wmma_gemm_kernel(const _Float16* __restrict__ A, const _Float16* __restrict__ Wpk,
                                 const float* __restrict__ bias, float* __restrict__ Y,
                                 float* __restrict__ sums, int Kp, int N) {
  int ntn    = N >> 4;               // 16-wide tiles along N
  int nquads = N >> 6;               // 64-wide super-tiles along N
  int q    = blockIdx.x % nquads;
  int m0   = blockIdx.x / nquads;
  int l    = threadIdx.x;            // 0..31, full wave, EXEC all ones
  int half = l >> 4;
  int lr   = l & 15;
  int n0   = q * 4;                  // first 16-wide tile of this super-tile
  v8f acc0 = {}, acc1 = {}, acc2 = {}, acc3 = {};
  // A-fragment: lanes 0-15 hold row lr, K = 0-7 & 16-23; lanes 16-31: K = 8-15 & 24-31
  const _Float16* arow  = A + (size_t)(m0 * 16 + lr) * Kp + 8 * half;
  const _Float16* wbase = Wpk + (((size_t)n0 * 32) + l) * 16;
  const size_t    wstep = (size_t)ntn * 512;   // advance one k-block (32 K)
  int kblocks = Kp >> 5;

  // prologue: fragments for kb = 0
  v16h a_cur = load_a_frag(arow, 0);
  v16h b0 = *(const v16h*)(wbase + 0 * 512);
  v16h b1 = *(const v16h*)(wbase + 1 * 512);
  v16h b2 = *(const v16h*)(wbase + 2 * 512);
  v16h b3 = *(const v16h*)(wbase + 3 * 512);

  for (int kb = 0; kb + 1 < kblocks; ++kb) {
    // issue k+1 loads first (overlap with the WMMAs below)
    const _Float16* wnext = wbase + (size_t)(kb + 1) * wstep;
    v16h a_n = load_a_frag(arow, kb + 1);
    v16h c0 = *(const v16h*)(wnext + 0 * 512);
    v16h c1 = *(const v16h*)(wnext + 1 * 512);
    v16h c2 = *(const v16h*)(wnext + 2 * 512);
    v16h c3 = *(const v16h*)(wnext + 3 * 512);
    acc0 = __builtin_amdgcn_wmma_f32_16x16x32_f16(false, a_cur, false, b0, (short)0, acc0, false, false);
    acc1 = __builtin_amdgcn_wmma_f32_16x16x32_f16(false, a_cur, false, b1, (short)0, acc1, false, false);
    acc2 = __builtin_amdgcn_wmma_f32_16x16x32_f16(false, a_cur, false, b2, (short)0, acc2, false, false);
    acc3 = __builtin_amdgcn_wmma_f32_16x16x32_f16(false, a_cur, false, b3, (short)0, acc3, false, false);
    a_cur = a_n; b0 = c0; b1 = c1; b2 = c2; b3 = c3;
  }
  // epilogue k-step
  acc0 = __builtin_amdgcn_wmma_f32_16x16x32_f16(false, a_cur, false, b0, (short)0, acc0, false, false);
  acc1 = __builtin_amdgcn_wmma_f32_16x16x32_f16(false, a_cur, false, b1, (short)0, acc1, false, false);
  acc2 = __builtin_amdgcn_wmma_f32_16x16x32_f16(false, a_cur, false, b2, (short)0, acc2, false, false);
  acc3 = __builtin_amdgcn_wmma_f32_16x16x32_f16(false, a_cur, false, b3, (short)0, acc3, false, false);

  v8f accs[4] = {acc0, acc1, acc2, acc3};
#pragma unroll
  for (int t = 0; t < 4; ++t) {      // c[v] -> M = v + 8*half, N = lr
    int col = (n0 + t) * 16 + lr;
    float bv = bias[col];
    float cs = 0.f, cq = 0.f;
#pragma unroll
    for (int v = 0; v < 8; ++v) {
      float cval = accs[t][v] + bv;
      int m = m0 * 16 + 8 * half + v;
      Y[(size_t)m * N + col] = cval;
      cs += cval; cq += cval * cval;
    }
    cs += __shfl_down(cs, 16, 32);   // fold half=1 into half=0 (same column)
    cq += __shfl_down(cq, 16, 32);
    if (half == 0) {
      atomicAdd(&sums[col], cs);
      atomicAdd(&sums[N + col], cq);
    }
  }
}

__global__ void bn_finalize_kernel(const float* __restrict__ sums, const float* __restrict__ g,
                                   const float* __restrict__ be, float* __restrict__ ss,
                                   int N, float invM) {
  int n = blockIdx.x * blockDim.x + threadIdx.x;
  if (n >= N) return;
  float mean = sums[n] * invM;
  float var  = sums[N + n] * invM - mean * mean;
  float sc   = g[n] * rsqrtf(var + BN_EPS);
  ss[n]     = sc;
  ss[N + n] = be[n] - mean * sc;
}

__global__ void bn_relu_kernel(const float* __restrict__ Y, const float* __restrict__ ss,
                               _Float16* __restrict__ X, int Nc, size_t total) {
  size_t i = (size_t)blockIdx.x * blockDim.x + threadIdx.x;
  if (i >= total) return;
  int n = (int)(i % Nc);
  float v = Y[i] * ss[n] + ss[Nc + n];
  X[i] = (_Float16)fmaxf(v, 0.f);
}

__global__ void maxpool_kernel(const _Float16* __restrict__ X, float* __restrict__ out,
                               int ns, int C) {
  int g = blockIdx.x;                // group (b*S + s)
  for (int c = threadIdx.x; c < C; c += blockDim.x) {
    float m = -3.4e38f;
    for (int j = 0; j < ns; ++j) {
      float v = (float)X[((size_t)g * ns + j) * C + c];
      if (v > m) m = v;
    }
    out[(size_t)g * C + c] = m;
  }
}

// ---------------- Tiny FC head (M = 16) ----------------
__global__ void fc_kernel(const float* __restrict__ X, const float* __restrict__ W,
                          const float* __restrict__ bias, float* __restrict__ Y,
                          int M, int K, int N) {
  int i = blockIdx.x * blockDim.x + threadIdx.x;
  if (i >= M * N) return;
  int m = i / N, n = i % N;
  float s = bias[n];
  for (int k = 0; k < K; ++k) s += X[(size_t)m * K + k] * W[(size_t)k * N + n];
  Y[i] = s;
}

__global__ void bn_rows_relu_kernel(float* __restrict__ Y, const float* __restrict__ g,
                                    const float* __restrict__ be, int M, int N) {
  int n = blockIdx.x * blockDim.x + threadIdx.x;
  if (n >= N) return;
  float s = 0.f, s2 = 0.f;
  for (int m = 0; m < M; ++m) { float v = Y[m * N + n]; s += v; s2 += v * v; }
  float mean = s / M;
  float var  = s2 / M - mean * mean;
  float sc   = g[n] * rsqrtf(var + BN_EPS);
  float sh   = be[n] - mean * sc;
  for (int m = 0; m < M; ++m) Y[m * N + n] = fmaxf(Y[m * N + n] * sc + sh, 0.f);
}

// ======================== host-side orchestration ========================
extern "C" void kernel_launch(void* const* d_in, const int* in_sizes, int n_in,
                              void* d_out, int out_size, void* d_ws, size_t ws_size,
                              hipStream_t stream) {
  (void)in_sizes; (void)n_in; (void)out_size; (void)ws_size;
  const int B = 16, N0 = 4096;
  const int S1 = 512, NS1 = 32;
  const int S2 = 128, NS2 = 64;

  // Inputs in setup_inputs() insertion order: points, then params
  // (sa1, sa2, sa3, fc1, fc2, fc3; each layer dict in W, b, g, be order).
  const float* points = (const float*)d_in[0];
  const float *saW[3][3], *sab[3][3], *sag[3][3], *sabe[3][3];
  int p = 1;
  for (int s = 0; s < 3; ++s)
    for (int L = 0; L < 3; ++L) {
      saW[s][L]  = (const float*)d_in[p++];
      sab[s][L]  = (const float*)d_in[p++];
      sag[s][L]  = (const float*)d_in[p++];
      sabe[s][L] = (const float*)d_in[p++];
    }
  const float* fc1W  = (const float*)d_in[p++];
  const float* fc1b  = (const float*)d_in[p++];
  const float* fc1g  = (const float*)d_in[p++];
  const float* fc1be = (const float*)d_in[p++];
  const float* fc2W  = (const float*)d_in[p++];
  const float* fc2b  = (const float*)d_in[p++];
  const float* fc2g  = (const float*)d_in[p++];
  const float* fc2be = (const float*)d_in[p++];
  const float* fc3W  = (const float*)d_in[p++];
  const float* fc3b  = (const float*)d_in[p++];

  // ---- workspace carve (256B aligned). Peak use ~265 MB. ----
  char* wp = (char*)d_ws;
  auto carve = [&](size_t bytes) -> void* {
    void* q = (void*)wp;
    wp += (bytes + 255) & ~(size_t)255;
    return q;
  };
  const size_t M1 = (size_t)B * S1 * NS1;   // 262144
  const size_t M2 = (size_t)B * S2 * NS2;   // 131072
  const size_t M3 = (size_t)B * S2;         // 2048
  _Float16* X0   = (_Float16*)carve(M1 * 128 * sizeof(_Float16));  // 64 MiB ping
  _Float16* X1   = (_Float16*)carve(M1 * 128 * sizeof(_Float16));  // 64 MiB pong
  float*    Ybuf = (float*)   carve(M1 * 128 * sizeof(float));     // 128 MiB preact
  _Float16* Wpk  = (_Float16*)carve((size_t)512 * 1024 * sizeof(_Float16));
  float*    sums = (float*)   carve(2 * 1024 * sizeof(float));
  float*    ss   = (float*)   carve(2 * 1024 * sizeof(float));
  int*      fidx1 = (int*)  carve((size_t)B * S1 * sizeof(int));
  float*    xyz1  = (float*)carve((size_t)B * S1 * 3 * sizeof(float));
  int*      gidx1 = (int*)  carve((size_t)B * S1 * NS1 * sizeof(int));
  int*      fidx2 = (int*)  carve((size_t)B * S2 * sizeof(int));
  float*    xyz2  = (float*)carve((size_t)B * S2 * 3 * sizeof(float));
  int*      gidx2 = (int*)  carve((size_t)B * S2 * NS2 * sizeof(int));
  float*    f1    = (float*)carve((size_t)B * S1 * 128 * sizeof(float));
  float*    f2    = (float*)carve((size_t)B * S2 * 256 * sizeof(float));
  float*    feat  = (float*)carve((size_t)B * 1024 * sizeof(float));
  float*    yfc1  = (float*)carve((size_t)B * 512 * sizeof(float));
  float*    yfc2  = (float*)carve((size_t)B * 256 * sizeof(float));

  auto run_layer = [&](const _Float16* Ain, int M, int K, int Kp, int N,
                       const float* Wm, const float* bm, const float* gm, const float* bem,
                       _Float16* Xout) {
    int pk_total = Kp * N;
    pack_w_kernel<<<(pk_total + 255) / 256, 256, 0, stream>>>(Wm, Wpk, K, Kp, N);
    zero_kernel<<<(2 * N + 255) / 256, 256, 0, stream>>>(sums, 2 * N);
    // 16x64 tile per wave; all N's here are multiples of 64.
    wmma_gemm_kernel<<<(M / 16) * (N / 64), 32, 0, stream>>>(Ain, Wpk, bm, Ybuf, sums, Kp, N);
    bn_finalize_kernel<<<(N + 255) / 256, 256, 0, stream>>>(sums, gm, bem, ss, N, 1.0f / (float)M);
    size_t total = (size_t)M * N;
    bn_relu_kernel<<<(unsigned)((total + 255) / 256), 256, 0, stream>>>(Ybuf, ss, Xout, N, total);
  };

  // ---------------- SA1 ----------------
  fps_kernel<<<B, 256, 0, stream>>>(points, fidx1, N0, S1);
  gather_xyz_kernel<<<(B * S1 + 255) / 256, 256, 0, stream>>>(points, fidx1, xyz1, N0, S1, B * S1);
  ball_query_kernel<<<B * S1, 32, 0, stream>>>(points, xyz1, gidx1, N0, S1, NS1, 0.2f * 0.2f);
  group_sa1_kernel<<<((int)M1 + 255) / 256, 256, 0, stream>>>(points, xyz1, gidx1, X0,
                                                              N0, S1, NS1, 32, (int)M1);
  run_layer(X0, (int)M1, 3,  32,  64, saW[0][0], sab[0][0], sag[0][0], sabe[0][0], X1);
  run_layer(X1, (int)M1, 64, 64,  64, saW[0][1], sab[0][1], sag[0][1], sabe[0][1], X0);
  run_layer(X0, (int)M1, 64, 64, 128, saW[0][2], sab[0][2], sag[0][2], sabe[0][2], X1);
  maxpool_kernel<<<B * S1, 256, 0, stream>>>(X1, f1, NS1, 128);

  // ---------------- SA2 ----------------
  fps_kernel<<<B, 256, 0, stream>>>(xyz1, fidx2, S1, S2);
  gather_xyz_kernel<<<(B * S2 + 255) / 256, 256, 0, stream>>>(xyz1, fidx2, xyz2, S1, S2, B * S2);
  ball_query_kernel<<<B * S2, 32, 0, stream>>>(xyz1, xyz2, gidx2, S1, S2, NS2, 0.4f * 0.4f);
  group_sa2_kernel<<<((int)M2 + 255) / 256, 256, 0, stream>>>(xyz1, xyz2, gidx2, f1, 128, X0,
                                                              S1, S2, NS2, 160, (int)M2);
  run_layer(X0, (int)M2, 131, 160, 128, saW[1][0], sab[1][0], sag[1][0], sabe[1][0], X1);
  run_layer(X1, (int)M2, 128, 128, 128, saW[1][1], sab[1][1], sag[1][1], sabe[1][1], X0);
  run_layer(X0, (int)M2, 128, 128, 256, saW[1][2], sab[1][2], sag[1][2], sabe[1][2], X1);
  maxpool_kernel<<<B * S2, 256, 0, stream>>>(X1, f2, NS2, 256);

  // ---------------- SA3 (group-all) ----------------
  group_sa3_kernel<<<((int)M3 + 255) / 256, 256, 0, stream>>>(xyz2, f2, 256, X0, 288, (int)M3);
  run_layer(X0, (int)M3, 259, 288,  256, saW[2][0], sab[2][0], sag[2][0], sabe[2][0], X1);
  run_layer(X1, (int)M3, 256, 256,  512, saW[2][1], sab[2][1], sag[2][1], sabe[2][1], X0);
  run_layer(X0, (int)M3, 512, 512, 1024, saW[2][2], sab[2][2], sag[2][2], sabe[2][2], X1);
  maxpool_kernel<<<B, 256, 0, stream>>>(X1, feat, S2, 1024);

  // ---------------- FC head ----------------
  fc_kernel<<<(B * 512 + 255) / 256, 256, 0, stream>>>(feat, fc1W, fc1b, yfc1, B, 1024, 512);
  bn_rows_relu_kernel<<<(512 + 255) / 256, 256, 0, stream>>>(yfc1, fc1g, fc1be, B, 512);
  fc_kernel<<<(B * 256 + 255) / 256, 256, 0, stream>>>(yfc1, fc2W, fc2b, yfc2, B, 512, 256);
  bn_rows_relu_kernel<<<(256 + 255) / 256, 256, 0, stream>>>(yfc2, fc2g, fc2be, B, 256);
  fc_kernel<<<(B * 40 + 255) / 256, 256, 0, stream>>>(yfc2, fc3W, fc3b, (float*)d_out, B, 256, 40);
}
